// InvariantPointAttention_34342558499034
// MI455X (gfx1250) — compile-verified, matrix operand order checked by reading, not run on hardware
//
#include <hip/hip_runtime.h>
#include <hip/hip_bf16.h>

// ---------------- problem constants ----------------
#define B_   2
#define N_   512
#define DIM_ 384
#define H_   8
#define PROJ_LD 672       // proj cols: sq[0,128) sk[128,256) sv[256,384) pq[384,480) pk[480,576) pv[576,672)
#define OUT_IN  3328      // 128 + 96 + 32 + 3072

static constexpr float SCALAR_SCALE = 0.14433756729740643f;   // (3*16)^-0.5
static constexpr float POINT_SCALE  = 0.13608276348795434f;   // (3*4*4.5)^-0.5
static constexpr float PAIR_SCALE   = 0.57735026918962576f;   // 3^-0.5

// ---------------- bf16 / WMMA plumbing ----------------
typedef __bf16 bf16_t;
typedef __attribute__((ext_vector_type(16))) bf16_t v16bf;
typedef __attribute__((ext_vector_type(2)))  bf16_t v2bf;
typedef __attribute__((ext_vector_type(8)))  float  v8f;
typedef __attribute__((ext_vector_type(4)))  float  v4f;
typedef __attribute__((ext_vector_type(8)))  unsigned short v8us;

union Frag {
  v16bf v;
  unsigned short u[16];
  v8us w[2];              // two 16-byte chunks (e=0..7, e=8..15)
  unsigned int d[8];
};

__device__ __forceinline__ unsigned short f2bf(float f) {
  unsigned int u = __float_as_uint(f);
  u += 0x7FFFu + ((u >> 16) & 1u);      // round-to-nearest-even
  return (unsigned short)(u >> 16);
}
__device__ __forceinline__ float bf2f(unsigned short h) {
  return __uint_as_float(((unsigned int)h) << 16);
}

#if __has_builtin(__builtin_amdgcn_cvt_pk_bf16_f32)
// single-instruction packed f32x2 -> bf16x2 conversion
__device__ __forceinline__ unsigned int pack2bf(float a, float b) {
  union { v2bf v; unsigned int u; } c;
  c.v = __builtin_amdgcn_cvt_pk_bf16_f32(a, b);
  return c.u;
}
#else
__device__ __forceinline__ unsigned int pack2bf(float a, float b) {
  return (unsigned int)f2bf(a) | ((unsigned int)f2bf(b) << 16);
}
#endif

// element e of a 16-bit A/B fragment maps to K = 8*half + e + (e>=8 ? 8 : 0)
__device__ __forceinline__ int frag_k(int e, int half) {
  return 8 * half + e + ((e >= 8) ? 8 : 0);
}
__device__ __forceinline__ v8f wmma_bf16(const Frag& a, const Frag& b, v8f c) {
  return __builtin_amdgcn_wmma_f32_16x16x32_bf16(false, a.v, false, b.v,
                                                 (short)0, c, false, false);
}
// load a fragment from bf16 memory where this lane's K-run is contiguous
template <typename P>
__device__ __forceinline__ void load_frag_bf16(Frag& f, const P* base, int half) {
  f.w[0] = *(const v8us*)(base + 8 * half);
  f.w[1] = *(const v8us*)(base + 8 * half + 16);
}

// ---------------- elementwise conversion kernels ----------------
__global__ void f32_to_bf16_kernel(const float* __restrict__ src,
                                   unsigned short* __restrict__ dst, long n) {
  long i = (long)blockIdx.x * blockDim.x + threadIdx.x;
  if (i < n) dst[i] = f2bf(src[i]);
}

// src (K x N) f32 -> dst (N x K) bf16  (transposed so B-fragments read contiguous K)
__global__ void f32_to_bf16_T_kernel(const float* __restrict__ src,
                                     unsigned short* __restrict__ dst, int K, int N) {
  int i = blockIdx.x * blockDim.x + threadIdx.x;
  if (i >= K * N) return;
  int k = i / N, n = i % N;
  dst[(long)n * K + k] = f2bf(src[i]);
}

// Wpair (384x8) -> bf16 transposed+padded (16 x 384), rows 8..15 zero
__global__ void pack_wpairT_kernel(const float* __restrict__ w,
                                   unsigned short* __restrict__ dst) {
  int i = blockIdx.x * blockDim.x + threadIdx.x;
  if (i >= 16 * DIM_) return;
  int h = i / DIM_, k = i % DIM_;
  dst[i] = (h < 8) ? f2bf(w[k * 8 + h]) : (unsigned short)0;
}

// ---------------- generic bf16 WMMA GEMM: C = A(MxK) * B^T(NxK) (+bias) ----------------
// block = 256 threads = 8 waves; wave w owns C tile (blockIdx.y, blockIdx.x*8+w)
// requires: M % 16 == 0, K % 32 == 0, N % 16 == 0
__global__ void gemm_bf16_kernel(const unsigned short* __restrict__ A,
                                 const unsigned short* __restrict__ BT,
                                 const float* __restrict__ bias,
                                 float* __restrict__ C,
                                 int M, int N, int K, int lda, int ldc) {
  const int lane = threadIdx.x & 31;
  const int wave = threadIdx.x >> 5;
  const int half = lane >> 4;
  const int r    = lane & 15;
  const int n0   = (blockIdx.x * 8 + wave) * 16;
  const int m0   = blockIdx.y * 16;
  if (n0 >= N) return;                       // wave-uniform

  const unsigned short* arow = A  + (long)(m0 + r) * lda;
  const unsigned short* brow = BT + (long)(n0 + r) * K;
  v8f acc = {};
  for (int k0 = 0; k0 < K; k0 += 32) {
    Frag a, b;
    load_frag_bf16(a, arow + k0, half);
    load_frag_bf16(b, brow + k0, half);
    acc = wmma_bf16(a, b, acc);
  }
  int n = n0 + r;
  float bs = bias ? bias[n] : 0.0f;
#pragma unroll
  for (int rr = 0; rr < 8; ++rr)
    C[(long)(m0 + rr + 8 * half) * ldc + n] = acc[rr] + bs;
}

// ---------------- points: rotate/translate q_p,k_p,v_p; |q|^2,|k|^2 ----------------
__global__ void points_kernel(const float* __restrict__ proj,
                              const float* __restrict__ rot,
                              const float* __restrict__ trans,
                              float* __restrict__ qp, float* __restrict__ kp,
                              float* __restrict__ vp,
                              float* __restrict__ qn, float* __restrict__ kn) {
  int idx = blockIdx.x * blockDim.x + threadIdx.x;     // (b,h,n) flat
  if (idx >= B_ * H_ * N_) return;
  int n = idx % N_;
  int h = (idx / N_) % H_;
  int b = idx / (N_ * H_);
  int row = b * N_ + n;
  const float* R = rot   + (long)row * 9;   // R[c][r]
  const float* t = trans + (long)row * 3;
  float qs = 0.f, ks = 0.f;
  for (int d = 0; d < 4; ++d) {
    float ql[3], kl[3], vl[3];
#pragma unroll
    for (int c = 0; c < 3; ++c) {
      long base = (long)row * PROJ_LD + h * 12 + d * 3 + c;
      ql[c] = proj[base + 384];
      kl[c] = proj[base + 480];
      vl[c] = proj[base + 576];
    }
#pragma unroll
    for (int rr = 0; rr < 3; ++rr) {
      float qg = t[rr], kg = t[rr], vg = t[rr];
#pragma unroll
      for (int c = 0; c < 3; ++c) {
        qg += ql[c] * R[c * 3 + rr];
        kg += kl[c] * R[c * 3 + rr];
        vg += vl[c] * R[c * 3 + rr];
      }
      long o = (long)idx * 12 + d * 3 + rr;
      qp[o] = qg; kp[o] = kg; vp[o] = vg;
      qs += qg * qg; ks += kg * kg;
    }
  }
  qn[idx] = qs; kn[idx] = ks;
}

// ---------------- pack attention operands: padded bf16, fragment-friendly ----------------
// qsb/ksb/qpb/kpb: [b][h][n][32]  (K padded with zeros: 16->32 / 12->32)
// vsT/vpT:         [b][h][16][512] (transposed, vpT rows 12..15 zero)
__global__ void pack_attn_kernel(const float* __restrict__ proj,
                                 const float* __restrict__ qp, const float* __restrict__ kp,
                                 const float* __restrict__ vp,
                                 unsigned short* __restrict__ qsb, unsigned short* __restrict__ ksb,
                                 unsigned short* __restrict__ qpb, unsigned short* __restrict__ kpb,
                                 unsigned short* __restrict__ vsT, unsigned short* __restrict__ vpT) {
  int idx = blockIdx.x * blockDim.x + threadIdx.x;     // (b,h,n)
  if (idx >= B_ * H_ * N_) return;
  int n = idx % N_;
  int h = (idx / N_) % H_;
  int b = idx / (N_ * H_);
  long row = (long)(b * N_ + n);
#pragma unroll
  for (int k = 0; k < 32; ++k) {
    qsb[(long)idx * 32 + k] = (k < 16) ? f2bf(proj[row * PROJ_LD + h * 16 + k])       : (unsigned short)0;
    ksb[(long)idx * 32 + k] = (k < 16) ? f2bf(proj[row * PROJ_LD + 128 + h * 16 + k]) : (unsigned short)0;
    qpb[(long)idx * 32 + k] = (k < 12) ? f2bf(qp[(long)idx * 12 + k])                 : (unsigned short)0;
    kpb[(long)idx * 32 + k] = (k < 12) ? f2bf(kp[(long)idx * 12 + k])                 : (unsigned short)0;
  }
  long hb = (long)(b * H_ + h);
#pragma unroll
  for (int d = 0; d < 16; ++d) {
    vsT[(hb * 16 + d) * N_ + n] = f2bf(proj[row * PROJ_LD + 256 + h * 16 + d]);
    vpT[(hb * 16 + d) * N_ + n] = (d < 12) ? f2bf(vp[(long)idx * 12 + d]) : (unsigned short)0;
  }
}

// ---------------- fused attention ----------------
// grid = B_*(N_/16), 512 threads (16 wave32s), dynamic LDS = 272KB
// Lb (bf16 attn/logits): [16 head-slots][16 i][512 j]; slots 8..15 zero (pads M for res_pair)
__global__ void __launch_bounds__(512, 1)
ipa_attention_kernel(const float* __restrict__ pairwise,
                     const unsigned short* __restrict__ qsb, const unsigned short* __restrict__ ksb,
                     const unsigned short* __restrict__ qpb, const unsigned short* __restrict__ kpb,
                     const unsigned short* __restrict__ vsT, const unsigned short* __restrict__ vpT,
                     const float* __restrict__ qn, const float* __restrict__ kn,
                     const unsigned short* __restrict__ wpairT,
                     const float* __restrict__ bpair,
                     const float* __restrict__ pweights,
                     const float* __restrict__ rot, const float* __restrict__ trans,
                     unsigned short* __restrict__ cat) {
  extern __shared__ unsigned short smem_us[];
  unsigned short* Lb = smem_us;                         // 16*16*512 bf16 (256KB)
  float* resS = (float*)(smem_us + 16 * 16 * N_);       // [8][16][16]
  float* resP = resS + 8 * 16 * 16;                     // [8][16][16]

  const int tid  = threadIdx.x;
  const int lane = tid & 31;
  const int wave = tid >> 5;                  // 0..15
  const int half = lane >> 4;
  const int r    = lane & 15;

  const int b  = blockIdx.x >> 5;
  const int i0 = (blockIdx.x & 31) * 16;

  // zero head-slots 8..15 (read as zero-padding by phase 2b A-fragments)
  for (int x = tid; x < 8 * 16 * N_; x += 512) Lb[8 * 16 * N_ + x] = 0;

  // ---- Phase 1a: scalar q.k + point q.k logits (wave -> head, j-half) ----
  {
    const int h = wave & 7;
    const int jhalf = wave >> 3;
    const float pw = log1pf(__expf(pweights[h]));
    const long hb = (long)(b * H_ + h);
    Frag qa, pa;
    load_frag_bf16(qa, qsb + (hb * N_ + i0 + r) * 32, half);
    load_frag_bf16(pa, qpb + (hb * N_ + i0 + r) * 32, half);
    float qn_m[8];
#pragma unroll
    for (int rr = 0; rr < 8; ++rr)
      qn_m[rr] = qn[hb * N_ + i0 + rr + 8 * half];

    for (int jt = jhalf * 16; jt < jhalf * 16 + 16; ++jt) {
      const int j0 = jt * 16;
      Frag kb, pb;
      load_frag_bf16(kb, ksb + (hb * N_ + j0 + r) * 32, half);
      load_frag_bf16(pb, kpb + (hb * N_ + j0 + r) * 32, half);
      v8f zero = {};
      v8f sAcc = wmma_bf16(qa, kb, zero);
      v8f pAcc = wmma_bf16(pa, pb, zero);
      float kn_j = kn[hb * N_ + j0 + r];
#pragma unroll
      for (int rr = 0; rr < 8; ++rr) {
        int m = rr + 8 * half;
        float dist = qn_m[rr] + kn_j - 2.0f * pAcc[rr];
        Lb[(h * 16 + m) * N_ + j0 + r] =
            f2bf(sAcc[rr] * SCALAR_SCALE - 0.5f * pw * POINT_SCALE * dist);
      }
    }
  }
  __syncthreads();

  // ---- Phase 1b: pair bias (wave -> i row); A from pairwise f32 (HBM read #1) ----
  {
    const int i = wave;
    const long prow = ((long)(b * N_ + i0 + i)) * N_;
    const float bp = (r < 8) ? bpair[r] : 0.0f;
    for (int jt = 0; jt < 32; ++jt) {
      const int j0 = jt * 16;
      if (jt + 1 < 32)
        __builtin_prefetch(&pairwise[(prow + j0 + 16 + r) * DIM_], 0, 1);
      v8f acc = {};
      for (int kc = 0; kc < 12; ++kc) {
        const int k0 = kc * 32;
        Frag a, wb;
        const float* ap = pairwise + (prow + j0 + r) * DIM_ + k0 + 8 * half;
        v4f x0 = *(const v4f*)(ap);
        v4f x1 = *(const v4f*)(ap + 4);
        v4f x2 = *(const v4f*)(ap + 16);
        v4f x3 = *(const v4f*)(ap + 20);
        a.d[0] = pack2bf(x0.x, x0.y); a.d[1] = pack2bf(x0.z, x0.w);
        a.d[2] = pack2bf(x1.x, x1.y); a.d[3] = pack2bf(x1.z, x1.w);
        a.d[4] = pack2bf(x2.x, x2.y); a.d[5] = pack2bf(x2.z, x2.w);
        a.d[6] = pack2bf(x3.x, x3.y); a.d[7] = pack2bf(x3.z, x3.w);
        load_frag_bf16(wb, wpairT + (long)r * DIM_ + k0, half);
        acc = wmma_bf16(a, wb, acc);
      }
      if (r < 8) {                       // col n = head
#pragma unroll
        for (int rr = 0; rr < 8; ++rr) {
          int jl = rr + 8 * half;
          unsigned short* p = &Lb[(r * 16 + i) * N_ + j0 + jl];
          *p = f2bf(bf2f(*p) + PAIR_SCALE * (acc[rr] + bp));
        }
      }
    }
  }
  __syncthreads();

  // ---- Softmax over j (wave -> head, row-half) ----
  {
    const int h = wave & 7;
    const int rbase = (wave >> 3) * 8;
    for (int m = rbase; m < rbase + 8; ++m) {
      unsigned short* row = &Lb[(h * 16 + m) * N_];
      float mx = -1e30f;
      for (int j = lane; j < N_; j += 32) mx = fmaxf(mx, bf2f(row[j]));
#pragma unroll
      for (int s = 16; s > 0; s >>= 1) mx = fmaxf(mx, __shfl_xor(mx, s, 32));
      float sum = 0.f;
      float ev[16];
#pragma unroll
      for (int t = 0; t < 16; ++t) {
        ev[t] = __expf(bf2f(row[lane + 32 * t]) - mx);
        sum += ev[t];
      }
#pragma unroll
      for (int s = 16; s > 0; s >>= 1) sum += __shfl_xor(sum, s, 32);
      float inv = 1.0f / sum;
#pragma unroll
      for (int t = 0; t < 16; ++t) row[lane + 32 * t] = f2bf(ev[t] * inv);
    }
  }
  __syncthreads();

  // ---- Phase 2a: res_scalar (part 0) / res_points (part 1), K = j = 512 ----
  {
    const int h = wave & 7;
    const int part = wave >> 3;
    const long hb = (long)(b * H_ + h);
    const unsigned short* vT = part ? vpT : vsT;
    v8f acc = {};
    for (int kt = 0; kt < 16; ++kt) {
      const int j0 = kt * 32;
      Frag a, vb;
      load_frag_bf16(a, &Lb[(h * 16 + r) * N_ + j0], half);       // ds_load_b128 x2
      load_frag_bf16(vb, vT + (hb * 16 + r) * N_ + j0, half);
      acc = wmma_bf16(a, vb, acc);
    }
    float* dst = part ? resP : resS;
#pragma unroll
    for (int rr = 0; rr < 8; ++rr)
      dst[(h * 16 + rr + 8 * half) * 16 + r] = acc[rr];
  }
  __syncthreads();

  // ---- Finalize scalar / rotate-back points / norms into concat (bf16) ----
  for (int x = tid; x < 2048; x += 512) {          // (h,i,d16) scalar part
    int d = x & 15, i = (x >> 4) & 15, h = x >> 8;
    long row = (long)(b * N_ + i0 + i);
    cat[row * OUT_IN + h * 16 + d] = f2bf(resS[(h * 16 + i) * 16 + d]);
  }
  {                                                 // (h,i,d4) point part: 512 items
    int x = tid;
    int d = x & 3, i = (x >> 2) & 15, h = x >> 6;
    long row = (long)(b * N_ + i0 + i);
    const float* R = rot   + row * 9;               // back-rotation uses R[r][c]
    const float* t = trans + row * 3;
    float loc[3];
#pragma unroll
    for (int rr = 0; rr < 3; ++rr) {
      float s = 0.f;
#pragma unroll
      for (int c = 0; c < 3; ++c)
        s += (resP[(h * 16 + i) * 16 + d * 3 + c] - t[c]) * R[rr * 3 + c];
      loc[rr] = s;
    }
    float nrm = sqrtf(loc[0]*loc[0] + loc[1]*loc[1] + loc[2]*loc[2] + 1e-8f);
#pragma unroll
    for (int c = 0; c < 3; ++c)
      cat[row * OUT_IN + 128 + h * 12 + d * 3 + c] = f2bf(loc[c]);
    cat[row * OUT_IN + 224 + h * 4 + d] = f2bf(nrm);
  }
  __syncthreads();

  // ---- Phase 2b: res_pair = attn(h,i,:) @ pairwise(i,:,:)  (L2-resident re-read) ----
  for (int task = wave; task < 384; task += 16) {   // 16 i x 24 d-tiles
    const int i  = task / 24;
    const int d0 = (task % 24) * 16;
    const long prow = ((long)(b * N_ + i0 + i)) * N_;
    v8f acc = {};
    for (int kt = 0; kt < 16; ++kt) {
      const int j0 = kt * 32;
      Frag a, pbg;
      load_frag_bf16(a, &Lb[(r * 16 + i) * N_ + j0], half);  // row m = head (slots 8..15 zero)
      float f[16];
#pragma unroll
      for (int e = 0; e < 16; ++e)
        f[e] = pairwise[(prow + j0 + frag_k(e, half)) * DIM_ + d0 + r];
#pragma unroll
      for (int t = 0; t < 8; ++t) pbg.d[t] = pack2bf(f[2 * t], f[2 * t + 1]);
      acc = wmma_bf16(a, pbg, acc);
    }
    long row = (long)(b * N_ + i0 + i);
    if (half == 0) {                                 // rows 0..7 = valid heads
#pragma unroll
      for (int rr = 0; rr < 8; ++rr)
        cat[row * OUT_IN + 256 + rr * DIM_ + d0 + r] = f2bf(acc[rr]);
    }
  }
}

// ---------------- host driver ----------------
extern "C" void kernel_launch(void* const* d_in, const int* in_sizes, int n_in,
                              void* d_out, int out_size, void* d_ws, size_t ws_size,
                              hipStream_t stream) {
  const float* x        = (const float*)d_in[0];
  const float* pairwise = (const float*)d_in[1];
  const float* rot      = (const float*)d_in[2];
  const float* trans    = (const float*)d_in[3];
  const float* Wsq      = (const float*)d_in[4];
  const float* Wsk      = (const float*)d_in[5];
  const float* Wsv      = (const float*)d_in[6];
  const float* Wpq      = (const float*)d_in[7];
  const float* Wpk      = (const float*)d_in[8];
  const float* Wpv      = (const float*)d_in[9];
  const float* pweights = (const float*)d_in[10];
  const float* Wpair    = (const float*)d_in[11];
  const float* bpair    = (const float*)d_in[12];
  const float* Wout     = (const float*)d_in[13];
  const float* bout     = (const float*)d_in[14];
  float* out = (float*)d_out;

  const int M = B_ * N_;                 // 1024
  const int BHN = B_ * H_ * N_;          // 8192
  char* ws = (char*)d_ws;
  size_t off = 0;
  auto alloc = [&](size_t bytes) { char* p = ws + off; off = (off + bytes + 255) & ~(size_t)255; return p; };

  unsigned short* xb      = (unsigned short*)alloc((size_t)M * DIM_ * 2);
  unsigned short* wsqT    = (unsigned short*)alloc((size_t)128 * DIM_ * 2);
  unsigned short* wskT    = (unsigned short*)alloc((size_t)128 * DIM_ * 2);
  unsigned short* wsvT    = (unsigned short*)alloc((size_t)128 * DIM_ * 2);
  unsigned short* wpqT    = (unsigned short*)alloc((size_t)96 * DIM_ * 2);
  unsigned short* wpkT    = (unsigned short*)alloc((size_t)96 * DIM_ * 2);
  unsigned short* wpvT    = (unsigned short*)alloc((size_t)96 * DIM_ * 2);
  unsigned short* wpairT  = (unsigned short*)alloc((size_t)16 * DIM_ * 2);
  unsigned short* woutT   = (unsigned short*)alloc((size_t)DIM_ * OUT_IN * 2);
  float* proj = (float*)alloc((size_t)M * PROJ_LD * 4);
  float* qp   = (float*)alloc((size_t)BHN * 12 * 4);
  float* kp   = (float*)alloc((size_t)BHN * 12 * 4);
  float* vp   = (float*)alloc((size_t)BHN * 12 * 4);
  float* qn   = (float*)alloc((size_t)BHN * 4);
  float* kn   = (float*)alloc((size_t)BHN * 4);
  unsigned short* qsb = (unsigned short*)alloc((size_t)BHN * 32 * 2);
  unsigned short* ksb = (unsigned short*)alloc((size_t)BHN * 32 * 2);
  unsigned short* qpb = (unsigned short*)alloc((size_t)BHN * 32 * 2);
  unsigned short* kpb = (unsigned short*)alloc((size_t)BHN * 32 * 2);
  unsigned short* vsT = (unsigned short*)alloc((size_t)B_ * H_ * 16 * N_ * 2);
  unsigned short* vpT = (unsigned short*)alloc((size_t)B_ * H_ * 16 * N_ * 2);
  unsigned short* cat = (unsigned short*)alloc((size_t)M * OUT_IN * 2);

  f32_to_bf16_kernel<<<dim3((M * DIM_ + 255) / 256), dim3(256), 0, stream>>>(x, xb, (long)M * DIM_);
  auto cvtT = [&](const float* s, unsigned short* d, int K, int N) {
    f32_to_bf16_T_kernel<<<dim3((K * N + 255) / 256), dim3(256), 0, stream>>>(s, d, K, N);
  };
  cvtT(Wsq, wsqT, DIM_, 128);
  cvtT(Wsk, wskT, DIM_, 128);
  cvtT(Wsv, wsvT, DIM_, 128);
  cvtT(Wpq, wpqT, DIM_, 96);
  cvtT(Wpk, wpkT, DIM_, 96);
  cvtT(Wpv, wpvT, DIM_, 96);
  cvtT(Wout, woutT, OUT_IN, DIM_);
  pack_wpairT_kernel<<<dim3((16 * DIM_ + 255) / 256), dim3(256), 0, stream>>>(Wpair, wpairT);

  // projections: proj[:, coff:coff+n] = x @ W
  struct { const unsigned short* wT; int n; int coff; } projs[6] = {
    { wsqT, 128, 0 }, { wskT, 128, 128 }, { wsvT, 128, 256 },
    { wpqT, 96, 384 }, { wpkT, 96, 480 }, { wpvT, 96, 576 } };
  for (int p = 0; p < 6; ++p) {
    dim3 g((projs[p].n + 127) / 128, M / 16);
    gemm_bf16_kernel<<<g, dim3(256), 0, stream>>>(
        xb, projs[p].wT, nullptr, proj + projs[p].coff,
        M, projs[p].n, DIM_, DIM_, PROJ_LD);
  }

  points_kernel<<<dim3((BHN + 255) / 256), dim3(256), 0, stream>>>(
      proj, rot, trans, qp, kp, vp, qn, kn);
  pack_attn_kernel<<<dim3((BHN + 255) / 256), dim3(256), 0, stream>>>(
      proj, qp, kp, vp, qsb, ksb, qpb, kpb, vsT, vpT);

  const int SMEM = 16 * 16 * N_ * 2 + 2 * (8 * 16 * 16) * 4;   // 272KB
  (void)hipFuncSetAttribute((const void*)ipa_attention_kernel,
                            hipFuncAttributeMaxDynamicSharedMemorySize, SMEM);
  ipa_attention_kernel<<<dim3(B_ * (N_ / 16)), dim3(512), SMEM, stream>>>(
      pairwise, qsb, ksb, qpb, kpb, vsT, vpT, qn, kn,
      wpairT, bpair, pweights, rot, trans, cat);

  // out = cat @ Wout + bout
  dim3 g((DIM_ + 127) / 128, M / 16);
  gemm_bf16_kernel<<<g, dim3(256), 0, stream>>>(
      cat, woutT, bout, out, M, DIM_, OUT_IN, OUT_IN, DIM_);
}
